// VectorNet_22625887715583
// MI455X (gfx1250) — compile-verified, matrix operand order checked by reading, not run on hardware
//
#include <hip/hip_runtime.h>
#include <hip/hip_bf16.h>

// ---------------------------------------------------------------------------
// VectorNet-style GATv2 pipeline for MI455X (gfx1250, wave32, WMMA).
//  - Dense projections: bf16 WMMA 16x16x32, fp32 accumulate.
//  - Edge phase: L2-resident gathers + native f32/i32 atomics (scatter softmax).
// ---------------------------------------------------------------------------

typedef __bf16 bf16;
typedef __attribute__((ext_vector_type(16))) bf16  v16bf;
typedef __attribute__((ext_vector_type(8)))  float v8f;

#define NNODES 50000
#define NEDGES 800000
#define NLANES 50000
#define NFOCAL 64
#define CLLEN  20

// ---------------------------------------------------------------------------
// Weight fragment prep: fp32 [Ci,Co] row-major -> bf16 WMMA B-fragments.
// Fragment order: frag[((kt*NT + nt)*32 + lane)*16 + i]
// B (32x16, 16-bit) lane layout mirrors the A layout from CDNA5 ISA:
//   lanes 0-15 hold column N=lane, K = {0..7,16..23}; lanes 16-31 K = {8..15,24..31}
// ---------------------------------------------------------------------------
__global__ void prep_frag(const float* __restrict__ W, bf16* __restrict__ frag,
                          int Ci, int Co)
{
    int KT = (Ci + 31) >> 5;
    int NT = Co >> 4;
    int total = KT * NT * 32 * 16;
    for (int idx = blockIdx.x * blockDim.x + threadIdx.x; idx < total;
         idx += gridDim.x * blockDim.x) {
        int i     = idx & 15;
        int lane  = (idx >> 4) & 31;
        int t     = idx >> 9;                 // kt*NT + nt
        int nt    = t % NT;
        int kt    = t / NT;
        int khalf = lane >> 4;
        int k     = kt * 32 + ((i < 8) ? (khalf * 8 + i) : (16 + khalf * 8 + (i - 8)));
        int ncol  = nt * 16 + (lane & 15);
        float v   = (k < Ci) ? W[(size_t)k * Co + ncol] : 0.0f;
        frag[idx] = (bf16)v;
    }
}

// ---------------------------------------------------------------------------
// GEMM: OUT[Nrows,Co] = act( A[Nrows,Ci] @ W (+ bias) )
// One wave = 16 output rows, loops over Co/16 column tiles; K tiled by 32.
// KT compile-time -> A-fragments live in registers; WMMA runs with full EXEC
// (row clamp on loads, masking only on the tail-tile store path).
// FULLK: Ci is a multiple of 32 -> unconditional float4 A loads.
// ---------------------------------------------------------------------------
template <int KT, bool FULLK, bool RELU>
__global__ void __launch_bounds__(128)
gemm_wmma(const float* __restrict__ A, const bf16* __restrict__ frag,
          const float* __restrict__ bias, float* __restrict__ OUT,
          int Nrows, int Ci, int Co)
{
    const int lane  = threadIdx.x & 31;
    const int wave  = threadIdx.x >> 5;
    const int row0  = (blockIdx.x * 4 + wave) * 16;
    const int khalf = lane >> 4;
    const int NT    = Co >> 4;

    int arow = row0 + (lane & 15);
    if (arow >= Nrows) arow = Nrows - 1;          // clamp (stores are masked)
    const float* ap = A + (size_t)arow * Ci;

    v16bf afrag[KT];
#pragma unroll
    for (int kt = 0; kt < KT; ++kt) {
        const int kbase = kt * 32 + khalf * 8;
        v16bf av;
        if (FULLK) {
            float4 f0 = *(const float4*)(ap + kbase);
            float4 f1 = *(const float4*)(ap + kbase + 4);
            float4 g0 = *(const float4*)(ap + kbase + 16);
            float4 g1 = *(const float4*)(ap + kbase + 20);
            av[0] = (bf16)f0.x;  av[1] = (bf16)f0.y;  av[2]  = (bf16)f0.z;  av[3]  = (bf16)f0.w;
            av[4] = (bf16)f1.x;  av[5] = (bf16)f1.y;  av[6]  = (bf16)f1.z;  av[7]  = (bf16)f1.w;
            av[8] = (bf16)g0.x;  av[9] = (bf16)g0.y;  av[10] = (bf16)g0.z;  av[11] = (bf16)g0.w;
            av[12] = (bf16)g1.x; av[13] = (bf16)g1.y; av[14] = (bf16)g1.z;  av[15] = (bf16)g1.w;
        } else {
#pragma unroll
            for (int i = 0; i < 8; ++i) {
                int k0 = kbase + i;
                int k1 = kbase + 16 + i;
                float f0 = (k0 < Ci) ? ap[k0] : 0.0f;
                float f1 = (k1 < Ci) ? ap[k1] : 0.0f;
                av[i]     = (bf16)f0;
                av[i + 8] = (bf16)f1;
            }
        }
        afrag[kt] = av;
    }

    const bool fullTile = (row0 + 16 <= Nrows);

    for (int nt = 0; nt < NT; ++nt) {
        v8f acc = {};
#pragma unroll
        for (int kt = 0; kt < KT; ++kt) {
            v16bf bv = *(const v16bf*)(frag + ((size_t)(kt * NT + nt) * 32 + lane) * 16);
            acc = __builtin_amdgcn_wmma_f32_16x16x32_bf16(
                      false, afrag[kt], false, bv, (short)0, acc, false, false);
        }
        // D layout: lanes 0-15 -> rows j, cols lane; lanes 16-31 -> rows j+8
        const int ncol  = nt * 16 + (lane & 15);
        const int mbase = row0 + ((lane >> 4) << 3);
        float bb = 0.0f;
        if (RELU) bb = bias[ncol];                 // hoisted: one load per tile
        float* op = OUT + (size_t)mbase * Co + ncol;
        if (fullTile) {
#pragma unroll
            for (int j = 0; j < 8; ++j) {
                float v = acc[j];
                if (RELU) { v += bb; v = v > 0.0f ? v : 0.0f; }
                op[(size_t)j * Co] = v;
            }
        } else {
#pragma unroll
            for (int j = 0; j < 8; ++j) {
                float v = acc[j];
                if (RELU) { v += bb; v = v > 0.0f ? v : 0.0f; }
                if (mbase + j < Nrows) op[(size_t)j * Co] = v;
            }
        }
    }
}

// ---------------------------------------------------------------------------
// Utility fills (d_ws is poisoned each run)
// ---------------------------------------------------------------------------
__global__ void fill_f32(float* p, float v, size_t n)
{
    for (size_t i = blockIdx.x * (size_t)blockDim.x + threadIdx.x; i < n;
         i += (size_t)gridDim.x * blockDim.x) p[i] = v;
}
__global__ void fill_i32(int* p, int v, size_t n)
{
    for (size_t i = blockIdx.x * (size_t)blockDim.x + threadIdx.x; i < n;
         i += (size_t)gridDim.x * blockDim.x) p[i] = v;
}

// Monotone float<->ordered-int map so segment-max can use native atomic_max_i32
__device__ __forceinline__ int f2ord(float f)
{
    int b = __float_as_int(f);
    return b >= 0 ? b : (b ^ 0x7fffffff);
}
__device__ __forceinline__ float ord2f(int b)
{
    return __int_as_float(b >= 0 ? b : (b ^ 0x7fffffff));
}

// ---------------------------------------------------------------------------
// Edge phase (one thread per (edge, head); C = 32 fixed, H compile-time)
// ---------------------------------------------------------------------------
template <int H>
__global__ void edge_score(const float* __restrict__ xl, const float* __restrict__ xr,
                           const float* __restrict__ att,
                           const int* __restrict__ srcI, const int* __restrict__ dstI,
                           int E, int EA,
                           float* __restrict__ score, int* __restrict__ smax)
{
    int idx = blockIdx.x * blockDim.x + threadIdx.x;
    if (idx >= EA * H) return;
    int e = idx / H, h = idx - e * H;
    int src = (e < E) ? srcI[e] : (e - E);
    int dst = (e < E) ? dstI[e] : (e - E);
    const float* pl = xl + ((size_t)src * H + h) * 32;
    const float* pr = xr + ((size_t)dst * H + h) * 32;
    const float* pa = att + h * 32;
    float s = 0.0f;
#pragma unroll
    for (int c = 0; c < 32; c += 4) {
        float4 a = *(const float4*)(pl + c);
        float4 b = *(const float4*)(pr + c);
        float4 w = *(const float4*)(pa + c);
        float v0 = a.x + b.x, v1 = a.y + b.y, v2 = a.z + b.z, v3 = a.w + b.w;
        v0 = v0 > 0.0f ? v0 : 0.2f * v0;
        v1 = v1 > 0.0f ? v1 : 0.2f * v1;
        v2 = v2 > 0.0f ? v2 : 0.2f * v2;
        v3 = v3 > 0.0f ? v3 : 0.2f * v3;
        s += w.x * v0 + w.y * v1 + w.z * v2 + w.w * v3;
    }
    score[idx] = s;
    atomicMax(smax + (size_t)dst * H + h, f2ord(s));
}

template <int H>
__global__ void edge_exp(float* __restrict__ sc, const int* __restrict__ smax,
                         const int* __restrict__ dstI, int E, int EA,
                         float* __restrict__ denom)
{
    int idx = blockIdx.x * blockDim.x + threadIdx.x;
    if (idx >= EA * H) return;
    int e = idx / H, h = idx - e * H;
    int dst = (e < E) ? dstI[e] : (e - E);
    float m  = ord2f(smax[(size_t)dst * H + h]);
    float ex = __expf(sc[idx] - m);
    sc[idx]  = ex;
    atomicAdd(denom + (size_t)dst * H + h, ex);
}

template <int H>
__global__ void edge_agg(const float* __restrict__ ex, const float* __restrict__ denom,
                         const float* __restrict__ xl,
                         const int* __restrict__ srcI, const int* __restrict__ dstI,
                         int E, int EA, float* __restrict__ agg)
{
    int idx = blockIdx.x * blockDim.x + threadIdx.x;
    if (idx >= EA * H) return;
    int e = idx / H, h = idx - e * H;
    int src = (e < E) ? srcI[e] : (e - E);
    int dst = (e < E) ? dstI[e] : (e - E);
    float alpha = ex[idx] / denom[(size_t)dst * H + h];
    const float* pl = xl + ((size_t)src * H + h) * 32;
    float* po = agg + ((size_t)dst * H + h) * 32;
#pragma unroll
    for (int c = 0; c < 32; c += 4) {
        float4 a = *(const float4*)(pl + c);
        atomicAdd(po + c + 0, alpha * a.x);
        atomicAdd(po + c + 1, alpha * a.y);
        atomicAdd(po + c + 2, alpha * a.z);
        atomicAdd(po + c + 3, alpha * a.w);
    }
}

template <int D>
__global__ void bias_relu(const float* __restrict__ in, const float* __restrict__ b,
                          float* __restrict__ out, size_t n)
{
    for (size_t i = blockIdx.x * (size_t)blockDim.x + threadIdx.x; i < n;
         i += (size_t)gridDim.x * blockDim.x) {
        float v = in[i] + b[i & (D - 1)];
        out[i] = v > 0.0f ? v : 0.0f;
    }
}

// ---------------------------------------------------------------------------
// Map branch: sum over rows of relu(lane_x @ map_W + map_b); divided by M later
// ---------------------------------------------------------------------------
__global__ void map_pool(const float* __restrict__ lane_x, const float* __restrict__ W,
                         const float* __restrict__ b, float* __restrict__ accum, int M)
{
    int t = blockIdx.x * blockDim.x + threadIdx.x;
    int c = t & 31;
    int rowStart  = t >> 5;
    int rowStride = (gridDim.x * blockDim.x) >> 5;
    float w0 = W[c], w1 = W[32 + c], bb = b[c];
    float s = 0.0f;
    for (int r = rowStart; r < M; r += rowStride) {
        float v = lane_x[2 * r] * w0 + lane_x[2 * r + 1] * w1 + bb;
        s += v > 0.0f ? v : 0.0f;
    }
    atomicAdd(accum + c, s);
}

// ---------------------------------------------------------------------------
// Head MLP over F=64 focal agents (tiny; one thread per agent)
// ---------------------------------------------------------------------------
__global__ void head_mlp(const float* __restrict__ gf, const int* __restrict__ focal,
                         const float* __restrict__ centerline,
                         const float* __restrict__ mapAcc, float invM,
                         const float* __restrict__ cl_W, const float* __restrict__ cl_b,
                         const float* __restrict__ fc1_W, const float* __restrict__ fc1_b,
                         const float* __restrict__ fc2_W, const float* __restrict__ fc2_b,
                         const float* __restrict__ fco_W, const float* __restrict__ fco_b,
                         float* __restrict__ out, int F)
{
    int f = blockIdx.x * blockDim.x + threadIdx.x;
    if (f >= F) return;
    float comb[96];
    const float* g = gf + (size_t)focal[f] * 32;
    for (int c = 0; c < 32; ++c) comb[c] = g[c];
    for (int c = 0; c < 32; ++c) comb[32 + c] = mapAcc[c] * invM;
    float mx = 0.0f, my = 0.0f;
    const float* cp = centerline + (size_t)f * (CLLEN * 2);
    for (int l = 0; l < CLLEN; ++l) { mx += cp[2 * l]; my += cp[2 * l + 1]; }
    mx *= (1.0f / CLLEN); my *= (1.0f / CLLEN);
    for (int c = 0; c < 32; ++c) comb[64 + c] = mx * cl_W[c] + my * cl_W[32 + c] + cl_b[c];
    float h1[32];
    for (int c = 0; c < 32; ++c) {
        float s = fc1_b[c];
        for (int j = 0; j < 96; ++j) s += comb[j] * fc1_W[j * 32 + c];
        h1[c] = fmaxf(s, 0.0f);
    }
    float h2[16];
    for (int c = 0; c < 16; ++c) {
        float s = fc2_b[c];
        for (int j = 0; j < 32; ++j) s += h1[j] * fc2_W[j * 16 + c];
        h2[c] = fmaxf(s, 0.0f);
    }
    for (int o = 0; o < 60; ++o) {
        float s = fco_b[o];
        for (int j = 0; j < 16; ++j) s += h2[j] * fco_W[j * 60 + o];
        out[(size_t)f * 60 + o] = s;
    }
}

// ---------------------------------------------------------------------------
extern "C" void kernel_launch(void* const* d_in, const int* in_sizes, int n_in,
                              void* d_out, int out_size, void* d_ws, size_t ws_size,
                              hipStream_t stream)
{
    (void)in_sizes; (void)n_in; (void)out_size; (void)ws_size;
    const int N = NNODES, E = NEDGES, M = NLANES, F = NFOCAL;
    const int EA = E + N;

    const float* x        = (const float*)d_in[0];
    const float* lane_x   = (const float*)d_in[1];
    const float* centerln = (const float*)d_in[2];
    const int*   eidx     = (const int*)d_in[3];
    const int*   focal    = (const int*)d_in[4];
    const float* emb_W    = (const float*)d_in[5];
    const float* emb_b    = (const float*)d_in[6];
    const float* c1_Wl    = (const float*)d_in[7];
    const float* c1_Wr    = (const float*)d_in[8];
    const float* c1_att   = (const float*)d_in[9];
    const float* c1_b     = (const float*)d_in[10];
    const float* c2_Wl    = (const float*)d_in[11];
    const float* c2_Wr    = (const float*)d_in[12];
    const float* c2_att   = (const float*)d_in[13];
    const float* c2_b     = (const float*)d_in[14];
    const float* map_W    = (const float*)d_in[15];
    const float* map_b    = (const float*)d_in[16];
    const float* gg_Wl    = (const float*)d_in[17];
    const float* gg_Wr    = (const float*)d_in[18];
    const float* gg_att   = (const float*)d_in[19];
    const float* gg_b     = (const float*)d_in[20];
    const float* cl_W     = (const float*)d_in[21];
    const float* cl_b     = (const float*)d_in[22];
    const float* fc1_W    = (const float*)d_in[23];
    const float* fc1_b    = (const float*)d_in[24];
    const float* fc2_W    = (const float*)d_in[25];
    const float* fc2_b    = (const float*)d_in[26];
    const float* fco_W    = (const float*)d_in[27];
    const float* fco_b    = (const float*)d_in[28];

    const int* srcI = eidx;
    const int* dstI = eidx + E;

    // ---- workspace carve (aligned 256B) -----------------------------------
    char*  ws  = (char*)d_ws;
    size_t off = 0;
    auto alloc = [&](size_t bytes) -> void* {
        void* p = (void*)(ws + off);
        off += (bytes + 255) & ~(size_t)255;
        return p;
    };
    float* bufA   = (float*)alloc((size_t)N * 128 * 4);   // node features (ping)
    float* bufXL  = (float*)alloc((size_t)N * 128 * 4);
    float* bufXR  = (float*)alloc((size_t)N * 128 * 4);
    float* bufAgg = (float*)alloc((size_t)N * 128 * 4);
    float* score  = (float*)alloc((size_t)EA * 4 * 4);
    int*   smax   = (int*)  alloc((size_t)N * 4 * 4);
    float* denom  = (float*)alloc((size_t)N * 4 * 4);
    float* gfbuf  = (float*)alloc((size_t)N * 32 * 4);
    float* mapAcc = (float*)alloc(32 * 4);
    bf16*  fEmb   = (bf16*)alloc(2048  * sizeof(bf16));   // KT=2, NT=2
    bf16*  fC1l   = (bf16*)alloc(4096  * sizeof(bf16));   // KT=1, NT=8
    bf16*  fC1r   = (bf16*)alloc(4096  * sizeof(bf16));
    bf16*  fC2l   = (bf16*)alloc(16384 * sizeof(bf16));   // KT=4, NT=8
    bf16*  fC2r   = (bf16*)alloc(16384 * sizeof(bf16));
    bf16*  fGGl   = (bf16*)alloc(4096  * sizeof(bf16));   // KT=4, NT=2
    bf16*  fGGr   = (bf16*)alloc(4096  * sizeof(bf16));

    // ---- weight fragment prep (tiny) --------------------------------------
    prep_frag<<<8,  256, 0, stream>>>(emb_W, fEmb, 60, 32);
    prep_frag<<<16, 256, 0, stream>>>(c1_Wl, fC1l, 32, 128);
    prep_frag<<<16, 256, 0, stream>>>(c1_Wr, fC1r, 32, 128);
    prep_frag<<<64, 256, 0, stream>>>(c2_Wl, fC2l, 128, 128);
    prep_frag<<<64, 256, 0, stream>>>(c2_Wr, fC2r, 128, 128);
    prep_frag<<<16, 256, 0, stream>>>(gg_Wl, fGGl, 128, 32);
    prep_frag<<<16, 256, 0, stream>>>(gg_Wr, fGGr, 128, 32);

    // ---- map branch --------------------------------------------------------
    fill_f32<<<1, 32, 0, stream>>>(mapAcc, 0.0f, 32);
    map_pool<<<64, 256, 0, stream>>>(lane_x, map_W, map_b, mapAcc, M);

    const int gGrid = (N + 63) / 64;   // 4 waves/block, 16 rows/wave

    // ---- agent embedding: bufA = relu(x @ emb_W + emb_b) -------------------
    gemm_wmma<2, false, true><<<gGrid, 128, 0, stream>>>(x, fEmb, emb_b, bufA, N, 60, 32);

    // ---- GATv2 layer runner ------------------------------------------------
    auto run_gat = [&](const float* inBuf, int Ci, const bf16* fWl, const bf16* fWr,
                       const float* att, const float* bvec, int H, float* outBuf) {
        const int Co = H * 32;
        if (Ci == 32) {
            gemm_wmma<1, true, false><<<gGrid, 128, 0, stream>>>(inBuf, fWl, nullptr, bufXL, N, Ci, Co);
            gemm_wmma<1, true, false><<<gGrid, 128, 0, stream>>>(inBuf, fWr, nullptr, bufXR, N, Ci, Co);
        } else {
            gemm_wmma<4, true, false><<<gGrid, 128, 0, stream>>>(inBuf, fWl, nullptr, bufXL, N, Ci, Co);
            gemm_wmma<4, true, false><<<gGrid, 128, 0, stream>>>(inBuf, fWr, nullptr, bufXR, N, Ci, Co);
        }
        fill_i32<<<(N * H + 255) / 256, 256, 0, stream>>>(smax, (int)0x80000000, (size_t)N * H);
        fill_f32<<<(N * H + 255) / 256, 256, 0, stream>>>(denom, 0.0f, (size_t)N * H);
        fill_f32<<<2048, 256, 0, stream>>>(bufAgg, 0.0f, (size_t)N * Co);
        const int eT = EA * H, eG = (eT + 255) / 256;
        if (H == 4) {
            edge_score<4><<<eG, 256, 0, stream>>>(bufXL, bufXR, att, srcI, dstI, E, EA, score, smax);
            edge_exp<4>  <<<eG, 256, 0, stream>>>(score, smax, dstI, E, EA, denom);
            edge_agg<4>  <<<eG, 256, 0, stream>>>(score, denom, bufXL, srcI, dstI, E, EA, bufAgg);
            bias_relu<128><<<2048, 256, 0, stream>>>(bufAgg, bvec, outBuf, (size_t)N * Co);
        } else {
            edge_score<1><<<eG, 256, 0, stream>>>(bufXL, bufXR, att, srcI, dstI, E, EA, score, smax);
            edge_exp<1>  <<<eG, 256, 0, stream>>>(score, smax, dstI, E, EA, denom);
            edge_agg<1>  <<<eG, 256, 0, stream>>>(score, denom, bufXL, srcI, dstI, E, EA, bufAgg);
            bias_relu<32><<<2048, 256, 0, stream>>>(bufAgg, bvec, outBuf, (size_t)N * Co);
        }
    };

    run_gat(bufA, 32,  fC1l, fC1r, c1_att, c1_b, 4, bufA);   // layer 1 -> [N,128]
    run_gat(bufA, 128, fC2l, fC2r, c2_att, c2_b, 4, bufA);   // layer 2 -> [N,128]
    run_gat(bufA, 128, fGGl, fGGr, gg_att, gg_b, 1, gfbuf);  // global  -> [N,32]

    // ---- head --------------------------------------------------------------
    head_mlp<<<(F + 63) / 64, 64, 0, stream>>>(
        gfbuf, focal, centerln, mapAcc, 1.0f / (float)M,
        cl_W, cl_b, fc1_W, fc1_b, fc2_W, fc2_b, fco_W, fco_b,
        (float*)d_out, F);
}
